// SmartBinaryLinear_29918742184793
// MI455X (gfx1250) — compile-verified, matrix operand order checked by reading, not run on hardware
//
#include <hip/hip_runtime.h>
#include <hip/hip_fp16.h>

// ---------------------------------------------------------------------------
// SmartBinaryLinear for MI455X (gfx1250, wave32, WMMA)
//   y[m,n] = sum_k x[m,k] * Wmix[n,k] + bias[n]
//   Wmix   = salient ? W : alpha[k] * sign(W)
// Pass 1: Wmix -> f16 (workspace), x -> f16 (workspace).
// Pass 2: 128x256 block-tile GEMM, 8 waves, wave tile 64x64.
//   TILE_K=64: 32 WMMAs per barrier pair per wave.
//   Double-buffered LDS fed by GLOBAL_LOAD_ASYNC_TO_LDS_B128 (ASYNCcnt),
//   f32 accumulation via v_wmma_f32_16x16x32_f16.
// ---------------------------------------------------------------------------

typedef __attribute__((ext_vector_type(16))) _Float16 v16h;
typedef __attribute__((ext_vector_type(8)))  _Float16 v8h;
typedef __attribute__((ext_vector_type(8)))  float    v8f;
typedef __attribute__((ext_vector_type(4)))  float    v4f;

#define D_IN   4096
#define D_OUT  11008
#define M_TOK  4096          // BATCH * SEQ = 2 * 2048

#define TILE_M 128           // block tile M (2 waves x 64)
#define TILE_N 256           // block tile N (4 waves x 64)
#define TILE_K 64            // two WMMA K steps per barrier pair

// ---------------------------------------------------------------------------
// x (f32) -> f16, 8 elements per thread
// ---------------------------------------------------------------------------
__global__ __launch_bounds__(256) void cvt_x_kernel(const float* __restrict__ x,
                                                    _Float16* __restrict__ xh) {
  long long i = ((long long)blockIdx.x * blockDim.x + threadIdx.x) * 8;
  v4f a = *(const v4f*)(x + i);
  v4f b = *(const v4f*)(x + i + 4);
  v8h o;
  o[0] = (_Float16)a[0]; o[1] = (_Float16)a[1];
  o[2] = (_Float16)a[2]; o[3] = (_Float16)a[3];
  o[4] = (_Float16)b[0]; o[5] = (_Float16)b[1];
  o[6] = (_Float16)b[2]; o[7] = (_Float16)b[3];
  *(v8h*)(xh + i) = o;
}

// ---------------------------------------------------------------------------
// Wmix (f16) = mask ? W : alpha[k]*sign(W); 8 elements (same row) per thread.
// ---------------------------------------------------------------------------
__global__ __launch_bounds__(256) void binarize_w_kernel(const float* __restrict__ w,
                                                         const unsigned char* __restrict__ mask,
                                                         const float* __restrict__ alpha,
                                                         _Float16* __restrict__ wh) {
  long long i = ((long long)blockIdx.x * blockDim.x + threadIdx.x) * 8;
  int kcol = (int)(i & (D_IN - 1));
  v8h o;
#pragma unroll
  for (int r = 0; r < 8; ++r) {
    float wv = w[i + r];
    float av = alpha[kcol + r];
    float bin = (wv > 0.0f) ? av : ((wv < 0.0f) ? -av : 0.0f);
    float vv  = mask[i + r] ? wv : bin;
    o[r] = (_Float16)vv;
  }
  *(v8h*)(wh + i) = o;
}

// ---------------------------------------------------------------------------
// Async global->LDS 16-byte copy (CDNA5 GLOBAL_LOAD_ASYNC_TO_LDS_B128).
// lds_off: logical LDS byte address (flat shared addr low 32 bits).
// ---------------------------------------------------------------------------
__device__ __forceinline__ void async_copy_b128(unsigned lds_off,
                                                const _Float16* gptr) {
  asm volatile("global_load_async_to_lds_b128 %0, %1, off"
               :: "v"(lds_off), "v"((unsigned long long)(size_t)gptr)
               : "memory");
}

// Issue one K-tile (A: 128x64, B: 256x64 halves) = 12 async b128 per thread.
__device__ __forceinline__ void issue_tile(const _Float16* __restrict__ A,
                                           const _Float16* __restrict__ Bw,
                                           _Float16* AsBuf, _Float16* BsBuf,
                                           int m0, int n0, int k0, int t) {
#pragma unroll
  for (int p = 0; p < 4; ++p) {            // A: 1024 chunks of 16B
    int chunk = t + 256 * p;
    int row = chunk >> 3, piece = chunk & 7;
    async_copy_b128((unsigned)(size_t)&AsBuf[row * TILE_K + piece * 8],
                    A + (size_t)(m0 + row) * D_IN + k0 + piece * 8);
  }
#pragma unroll
  for (int p = 0; p < 8; ++p) {            // B: 2048 chunks of 16B
    int chunk = t + 256 * p;
    int row = chunk >> 3, piece = chunk & 7;
    async_copy_b128((unsigned)(size_t)&BsBuf[row * TILE_K + piece * 8],
                    Bw + (size_t)(n0 + row) * D_IN + k0 + piece * 8);
  }
}

// ---------------------------------------------------------------------------
// Fragment load from an LDS tile stored row-major [rows][TILE_K] halves.
// 16-bit A/B layout (ISA 7.12.2): lane L holds row (L&15), K-chunks at
// ks + 8*(L>>4) and ks + 16 + 8*(L>>4), each chunk 8 contiguous halves (16B).
// ---------------------------------------------------------------------------
__device__ __forceinline__ v16h load_frag_lds(const _Float16* base, int row0,
                                              int ks, int lane) {
  int r  = row0 + (lane & 15);
  int kc = ks + (lane >> 4) * 8;
  union { v16h v; v8h h[2]; } u;
  u.h[0] = *(const v8h*)(base + r * TILE_K + kc);
  u.h[1] = *(const v8h*)(base + r * TILE_K + kc + 16);
  return u.v;
}

// ---------------------------------------------------------------------------
// GEMM: C[M_TOK, D_OUT] = Xh * Wh^T + bias.
// Block: 256 threads = 8 waves (2 M x 4 N), wave tile 64x64 = 4x4 WMMA.
// ---------------------------------------------------------------------------
__global__ __launch_bounds__(256) void gemm_wmma_kernel(const _Float16* __restrict__ A,   // [M_TOK, D_IN]
                                                        const _Float16* __restrict__ Bw,  // [D_OUT, D_IN]
                                                        const float* __restrict__ bias,   // [D_OUT]
                                                        float* __restrict__ C) {          // [M_TOK, D_OUT]
  __shared__ _Float16 As[2][TILE_M * TILE_K];   // 2 x 16 KB
  __shared__ _Float16 Bs[2][TILE_N * TILE_K];   // 2 x 32 KB

  const int tid  = threadIdx.x;
  const int lane = tid & 31;
  const int wave = tid >> 5;       // 0..7
  const int wm   = wave >> 2;      // 0..1  -> M sub-tile (64 rows)
  const int wn   = wave & 3;       // 0..3  -> N sub-tile (64 cols)

  const int m0 = blockIdx.y * TILE_M;
  const int n0 = blockIdx.x * TILE_N;

  v8f acc[4][4];
#pragma unroll
  for (int i = 0; i < 4; ++i)
#pragma unroll
    for (int j = 0; j < 4; ++j)
      acc[i][j] = (v8f){0.f, 0.f, 0.f, 0.f, 0.f, 0.f, 0.f, 0.f};

  const int NT = D_IN / TILE_K;    // 64 K-tiles

  // Prologue: async-stage tile 0 into buffer 0.
  issue_tile(A, Bw, &As[0][0], &Bs[0][0], m0, n0, 0, tid);

  for (int it = 0; it < NT; ++it) {
    const int cur = it & 1;

    if (it + 1 < NT) {
      // Stage next tile into the other buffer while we compute this one.
      issue_tile(A, Bw, &As[cur ^ 1][0], &Bs[cur ^ 1][0],
                 m0, n0, (it + 1) * TILE_K, tid);
      // In-order async completion: <=12 outstanding => current tile landed.
      asm volatile("s_wait_asynccnt 0xc" ::: "memory");
    } else {
      asm volatile("s_wait_asynccnt 0x0" ::: "memory");
    }
    __syncthreads();   // current tile visible to all waves

#pragma unroll
    for (int ks = 0; ks < TILE_K; ks += 32) {
      v16h afrag[4];
#pragma unroll
      for (int i = 0; i < 4; ++i)
        afrag[i] = load_frag_lds(&As[cur][0], wm * 64 + i * 16, ks, lane);

      v16h bfrag[4];
#pragma unroll
      for (int j = 0; j < 4; ++j)
        bfrag[j] = load_frag_lds(&Bs[cur][0], wn * 64 + j * 16, ks, lane);

#pragma unroll
      for (int i = 0; i < 4; ++i)
#pragma unroll
        for (int j = 0; j < 4; ++j)
          acc[i][j] = __builtin_amdgcn_wmma_f32_16x16x32_f16(
              /*neg_a=*/false, afrag[i],
              /*neg_b=*/false, bfrag[j],
              /*c_mod=*/(short)0, acc[i][j],
              /*reuse_a=*/false, /*reuse_b=*/false);
    }

    __syncthreads();   // everyone done reading buf[cur] before it's re-filled
  }

  // Epilogue. C/D layout: VGPR r = row (r + 8*(lane>>4)), col = lane&15.
  const int col    = lane & 15;
  const int rowsel = lane >> 4;
#pragma unroll
  for (int i = 0; i < 4; ++i) {
#pragma unroll
    for (int j = 0; j < 4; ++j) {
      int gm = m0 + wm * 64 + i * 16 + rowsel * 8;
      int gn = n0 + wn * 64 + j * 16 + col;
      float bv = bias[gn];
      float* outp = C + (size_t)gm * D_OUT + gn;
#pragma unroll
      for (int r = 0; r < 8; ++r)
        outp[(size_t)r * D_OUT] = acc[i][j][r] + bv;
    }
  }
}

// ---------------------------------------------------------------------------
// Launch
// ---------------------------------------------------------------------------
extern "C" void kernel_launch(void* const* d_in, const int* in_sizes, int n_in,
                              void* d_out, int out_size, void* d_ws, size_t ws_size,
                              hipStream_t stream) {
  const float*         x     = (const float*)d_in[0];          // [2,2048,4096]
  const float*         w     = (const float*)d_in[1];          // [11008,4096]
  const float*         bias  = (const float*)d_in[2];          // [11008]
  const float*         alpha = (const float*)d_in[3];          // [4096]
  const unsigned char* mask  = (const unsigned char*)d_in[4];  // [11008,4096] bool
  float*               out   = (float*)d_out;                  // [2,2048,11008]

  char* ws = (char*)d_ws;
  _Float16* xh = (_Float16*)ws;
  size_t x_bytes = ((size_t)M_TOK * D_IN * sizeof(_Float16) + 255) & ~(size_t)255;
  _Float16* wh = (_Float16*)(ws + x_bytes);

  // 1) x -> f16
  {
    long long nthreads = (long long)M_TOK * D_IN / 8;
    cvt_x_kernel<<<(unsigned)(nthreads / 256), 256, 0, stream>>>(x, xh);
  }

  // 2) Wmix -> f16
  {
    long long nthreads = (long long)D_OUT * D_IN / 8;
    binarize_w_kernel<<<(unsigned)(nthreads / 256), 256, 0, stream>>>(w, mask, alpha, wh);
  }

  // 3) GEMM + bias
  {
    dim3 grid(D_OUT / TILE_N, M_TOK / TILE_M);   // (43, 32)
    gemm_wmma_kernel<<<grid, 256, 0, stream>>>(xh, wh, bias, out);
  }
}
